// NONA_NN_51702816309500
// MI455X (gfx1250) — matrix operand order repaired and checked by reading, not compiled, not run on hardware
//
#include <hip/hip_runtime.h>
#include <hip/hip_bf16.h>

// ---------------------------------------------------------------------------
// Types for WMMA fragments (gfx1250, wave32)
// ---------------------------------------------------------------------------
typedef __attribute__((ext_vector_type(16))) __bf16 bf16x16;
typedef __attribute__((ext_vector_type(8)))  __bf16 bf16x8;
typedef __attribute__((ext_vector_type(8)))  float  f32x8;

union BFrag { bf16x16 v; bf16x8 h[2]; };

#define TILE_M 256
#define TILE_N 128
#define TILE_K 32
#define N_KEYS 16384
#define N_CLS  100
#define QSLAB  1024

// ---------------------------------------------------------------------------
// Helper kernels
// ---------------------------------------------------------------------------
__global__ void cvt_f32_to_bf16_k(const float* __restrict__ in,
                                  __bf16* __restrict__ out, int n) {
  int i = blockIdx.x * 256 + threadIdx.x;
  if (i < n) out[i] = (__bf16)in[i];
}

__global__ void zero_f32_k(float* __restrict__ p, int n) {
  int i = blockIdx.x * 256 + threadIdx.x;
  if (i < n) p[i] = 0.0f;
}

// Column-wise sum / sum-of-squares with split-K over blockIdx.y (atomics).
__global__ void bn_stats_k(const void* __restrict__ in, int is_f32,
                           float* __restrict__ sums, int M, int N) {
  int c = blockIdx.x * 256 + threadIdx.x;
  if (c >= N) return;
  int chunk = M / gridDim.y;
  int r0 = blockIdx.y * chunk;
  float s = 0.0f, s2 = 0.0f;
  if (is_f32) {
    const float* pf = (const float*)in;
    for (int r = r0; r < r0 + chunk; ++r) {
      float v = pf[(size_t)r * N + c];
      s += v; s2 += v * v;
    }
  } else {
    const __bf16* pb = (const __bf16*)in;
    for (int r = r0; r < r0 + chunk; ++r) {
      float v = (float)pb[(size_t)r * N + c];
      s += v; s2 += v * v;
    }
  }
  atomicAdd(&sums[c], s);
  atomicAdd(&sums[N + c], s2);
}

__global__ void bn_finalize_k(const float* __restrict__ sums,
                              float* __restrict__ mr, int M, int N) {
  int c = blockIdx.x * 256 + threadIdx.x;
  if (c >= N) return;
  float m   = sums[c] / (float)M;
  float var = sums[N + c] / (float)M - m * m;
  mr[c]     = m;
  mr[N + c] = rsqrtf(var + 1e-5f);
}

__global__ void bn_apply_k(const void* __restrict__ in, int is_f32,
                           const float* __restrict__ mr,
                           const float* __restrict__ g,
                           const float* __restrict__ bt,
                           __bf16* __restrict__ out, int M, int N) {
  size_t i = (size_t)blockIdx.x * 256 + threadIdx.x;
  size_t total = (size_t)M * N;
  if (i >= total) return;
  int c = (int)(i % (size_t)N);
  float v = is_f32 ? ((const float*)in)[i] : (float)((const __bf16*)in)[i];
  out[i] = (__bf16)(g[c] * (v - mr[c]) * mr[N + c] + bt[c]);
}

// Per-row sum of squares of bf16 matrix (cols = N), one block per row.
__global__ void row_sumsq_k(const __bf16* __restrict__ in,
                            float* __restrict__ out, int N) {
  __shared__ float red[128];
  int row = blockIdx.x;
  float s = 0.0f;
  for (int c = threadIdx.x; c < N; c += 128) {
    float v = (float)in[(size_t)row * N + c];
    s += v * v;
  }
  red[threadIdx.x] = s;
  __syncthreads();
  for (int st = 64; st > 0; st >>= 1) {
    if (threadIdx.x < st) red[threadIdx.x] += red[threadIdx.x + st];
    __syncthreads();
  }
  if (threadIdx.x == 0) out[row] = red[0];
}

// ---------------------------------------------------------------------------
// WMMA GEMM: C[M,N] = A[M,K] * B[N,K]^T  (both bf16 row-major, K contiguous)
// MODE 0: C_bf16 = tanh(acc + bias[n])
// MODE 1: C_f32  = -sqrt(max(a2[m] + b2[n] - 2*acc, 0))
// Tile 256x128x32, 256 threads = 8 waves (4x2), each wave computes 64x64
// (4x4 WMMA tiles -> 16 v_wmma per k-step from 16 ds_load_b128).
// ---------------------------------------------------------------------------
template <int MODE>
__global__ __launch_bounds__(256)
void gemm_wmma_k(const __bf16* __restrict__ A, const __bf16* __restrict__ B,
                 const float* __restrict__ bias,
                 const float* __restrict__ a2, const float* __restrict__ b2,
                 __bf16* __restrict__ Cb, float* __restrict__ Cf,
                 int M, int N, int K) {
  __shared__ __align__(16) __bf16 As[TILE_M][TILE_K + 8];
  __shared__ __align__(16) __bf16 Bs[TILE_N][TILE_K + 8];

  const int tid  = threadIdx.x;
  const int wave = tid >> 5;
  const int lane = tid & 31;
  const int lh   = lane >> 4;   // lane half (0/1)
  const int lr   = lane & 15;

  const int bm = blockIdx.y * TILE_M;
  const int bn = blockIdx.x * TILE_N;
  const int wm = (wave >> 1) * 64;   // 0,64,128,192
  const int wn = (wave & 1) * 64;    // 0,64

  f32x8 acc[4][4];
#pragma unroll
  for (int i = 0; i < 4; ++i)
#pragma unroll
    for (int j = 0; j < 4; ++j)
      acc[i][j] = f32x8{0.f, 0.f, 0.f, 0.f, 0.f, 0.f, 0.f, 0.f};

  const int lrow = tid >> 2;        // 0..63
  const int lseg = (tid & 3) * 8;   // 0,8,16,24

  for (int k0 = 0; k0 < K; k0 += TILE_K) {
    // Stage tiles (each thread: 8 contiguous bf16 = one b128 per pass)
#pragma unroll
    for (int p = 0; p < TILE_M / 64; ++p) {
      int r = lrow + p * 64;
      *(bf16x8*)&As[r][lseg] =
          *(const bf16x8*)&A[(size_t)(bm + r) * K + k0 + lseg];
    }
#pragma unroll
    for (int p = 0; p < TILE_N / 64; ++p) {
      int r = lrow + p * 64;
      *(bf16x8*)&Bs[r][lseg] =
          *(const bf16x8*)&B[(size_t)(bn + r) * K + k0 + lseg];
    }
    __syncthreads();

    // B fragment (32x16 bf16): lane n holds column N=n; lanes 0-15 K0..15,
    // lanes 16-31 K16..31.  Loaded once, reused by 4 M-tiles.
    BFrag bf[4];
#pragma unroll
    for (int nt = 0; nt < 4; ++nt) {
      int col = wn + nt * 16 + lr;
      bf[nt].h[0] = *(const bf16x8*)&Bs[col][lh * 16];
      bf[nt].h[1] = *(const bf16x8*)&Bs[col][lh * 16 + 8];
    }
    // A fragment (16x32 bf16): lanes 0-15 hold K0..15,
    // lanes 16-31 hold K16..31.
#pragma unroll
    for (int mt = 0; mt < 4; ++mt) {
      BFrag af;
      int row = wm + mt * 16 + lr;
      af.h[0] = *(const bf16x8*)&As[row][lh * 16];
      af.h[1] = *(const bf16x8*)&As[row][lh * 16 + 8];
#pragma unroll
      for (int nt = 0; nt < 4; ++nt)
        acc[mt][nt] = __builtin_amdgcn_wmma_f32_16x16x32_bf16(
            false, af.v, false, bf[nt].v, (short)0, acc[mt][nt],
            false, false);
    }
    __syncthreads();
  }

  // Epilogue. D layout: VGPR r -> M = r + lh*8, N = lr (within 16x16 tile).
#pragma unroll
  for (int mt = 0; mt < 4; ++mt) {
#pragma unroll
    for (int nt = 0; nt < 4; ++nt) {
      int colg = bn + wn + nt * 16 + lr;
      float cv = (MODE == 0) ? bias[colg] : b2[colg];
#pragma unroll
      for (int r = 0; r < 8; ++r) {
        int rowg = bm + wm + mt * 16 + lh * 8 + r;
        float v = acc[mt][nt][r];
        if (MODE == 0) {
          Cb[(size_t)rowg * N + colg] = (__bf16)tanhf(v + cv);
        } else {
          float d2 = a2[rowg] + cv - 2.0f * v;
          Cf[(size_t)rowg * N + colg] = -sqrtf(fmaxf(d2, 0.0f));
        }
      }
    }
  }
}

// ---------------------------------------------------------------------------
// Fused softmax over 16384 keys + one-hot class aggregation, one block/query.
// out[q][c] = clip( sum_{k: y==c} exp(sim-max) / sum_k exp(sim-max), 0, 1 )
// ---------------------------------------------------------------------------
__global__ __launch_bounds__(256)
void softmax_agg_k(const float* __restrict__ sim, const int* __restrict__ yn,
                   float* __restrict__ out) {
  __shared__ float red[256];
  __shared__ float bins[8][N_CLS];
  const int q   = blockIdx.x;
  const int tid = threadIdx.x;
  const int wv  = tid >> 5;
  const float* row = sim + (size_t)q * N_KEYS;

  for (int c = tid; c < 8 * N_CLS; c += 256) (&bins[0][0])[c] = 0.0f;

  float mx = -3.4e38f;
  for (int k = tid; k < N_KEYS; k += 256) mx = fmaxf(mx, row[k]);
  red[tid] = mx;
  __syncthreads();
  for (int st = 128; st > 0; st >>= 1) {
    if (tid < st) red[tid] = fmaxf(red[tid], red[tid + st]);
    __syncthreads();
  }
  mx = red[0];
  __syncthreads();

  for (int k = tid; k < N_KEYS; k += 256) {
    float e = __expf(row[k] - mx);
    atomicAdd(&bins[wv][yn[k]], e);
  }
  __syncthreads();

  for (int c = tid; c < N_CLS; c += 256) {
    float s = 0.0f;
#pragma unroll
    for (int w = 0; w < 8; ++w) s += bins[w][c];
    bins[0][c] = s;
  }
  __syncthreads();

  __shared__ float totsh;
  if (tid == 0) {
    float t = 0.0f;
    for (int c = 0; c < N_CLS; ++c) t += bins[0][c];
    totsh = t;
  }
  __syncthreads();
  float inv = 1.0f / totsh;
  for (int c = tid; c < N_CLS; c += 256)
    out[(size_t)q * N_CLS + c] = fminf(fmaxf(bins[0][c] * inv, 0.0f), 1.0f);
}

// ---------------------------------------------------------------------------
// Host-side orchestration
// ---------------------------------------------------------------------------
extern "C" void kernel_launch(void* const* d_in, const int* in_sizes, int n_in,
                              void* d_out, int out_size, void* d_ws,
                              size_t ws_size, hipStream_t stream) {
  (void)in_sizes; (void)n_in; (void)out_size; (void)ws_size;

  const float* x   = (const float*)d_in[0];   // [4096,512]
  const float* xn  = (const float*)d_in[1];   // [16384,512]
  const int*   yn  = (const int*)d_in[2];     // [16384]
  const float* ing = (const float*)d_in[3];
  const float* inb = (const float*)d_in[4];
  const float* W[3]    = {(const float*)d_in[5], (const float*)d_in[9],
                          (const float*)d_in[13]};
  const float* bvec[3] = {(const float*)d_in[6], (const float*)d_in[10],
                          (const float*)d_in[14]};
  const float* gam[3]  = {(const float*)d_in[7], (const float*)d_in[11],
                          (const float*)d_in[15]};
  const float* bet[3]  = {(const float*)d_in[8], (const float*)d_in[12],
                          (const float*)d_in[16]};

  const int dims[4] = {512, 1024, 1024, 512};
  const int MQ = 4096, MN = 16384;

  // Workspace carving (256B aligned)
  char* wsp = (char*)d_ws;
  auto carve = [&](size_t bytes) -> void* {
    void* r = (void*)wsp;
    wsp += (bytes + 255) & ~(size_t)255;
    return r;
  };
  __bf16* wb[3];
  wb[0] = (__bf16*)carve((size_t)1024 * 512 * 2);
  wb[1] = (__bf16*)carve((size_t)1024 * 1024 * 2);
  wb[2] = (__bf16*)carve((size_t)512 * 1024 * 2);
  __bf16* xq0 = (__bf16*)carve((size_t)MQ * 1024 * 2);
  __bf16* xq1 = (__bf16*)carve((size_t)MQ * 1024 * 2);
  __bf16* xn0 = (__bf16*)carve((size_t)MN * 1024 * 2);
  __bf16* xn1 = (__bf16*)carve((size_t)MN * 1024 * 2);
  float* sums = (float*)carve((size_t)2 * 1024 * 4);
  float* mr   = (float*)carve((size_t)2 * 1024 * 4);
  float* a2   = (float*)carve((size_t)MQ * 4);
  float* b2   = (float*)carve((size_t)MN * 4);
  float* sim  = (float*)carve((size_t)QSLAB * N_KEYS * 4);

  // Convert weights to bf16
  {
    int wsz[3] = {1024 * 512, 1024 * 1024, 512 * 1024};
    for (int i = 0; i < 3; ++i)
      cvt_f32_to_bf16_k<<<dim3((wsz[i] + 255) / 256), 256, 0, stream>>>(
          W[i], wb[i], wsz[i]);
  }

  auto run_bn = [&](const void* in, int is_f32, const float* g,
                    const float* bt, __bf16* outb, int M, int N) {
    zero_f32_k<<<dim3((2 * N + 255) / 256), 256, 0, stream>>>(sums, 2 * N);
    bn_stats_k<<<dim3((N + 255) / 256, 32), 256, 0, stream>>>(in, is_f32,
                                                              sums, M, N);
    bn_finalize_k<<<dim3((N + 255) / 256), 256, 0, stream>>>(sums, mr, M, N);
    size_t total = (size_t)M * N;
    bn_apply_k<<<dim3((unsigned)((total + 255) / 256)), 256, 0, stream>>>(
        in, is_f32, mr, g, bt, outb, M, N);
  };

  // Input BatchNorm (separate stats for x and x_n, shared gamma/beta)
  run_bn(x, 1, ing, inb, xq0, MQ, 512);
  run_bn(xn, 1, ing, inb, xn0, MN, 512);

  // MLP layers: GEMM(+bias+tanh) -> BN -> bf16 activations (ping-pong)
  for (int L = 0; L < 3; ++L) {
    int K = dims[L], N = dims[L + 1];
    gemm_wmma_k<0><<<dim3(N / TILE_N, MQ / TILE_M), 256, 0, stream>>>(
        xq0, wb[L], bvec[L], nullptr, nullptr, xq1, nullptr, MQ, N, K);
    run_bn(xq1, 0, gam[L], bet[L], xq0, MQ, N);
    gemm_wmma_k<0><<<dim3(N / TILE_N, MN / TILE_M), 256, 0, stream>>>(
        xn0, wb[L], bvec[L], nullptr, nullptr, xn1, nullptr, MN, N, K);
    run_bn(xn1, 0, gam[L], bet[L], xn0, MN, N);
  }

  // Row squared norms of final 512-d features
  row_sumsq_k<<<dim3(MQ), 128, 0, stream>>>(xq0, a2, 512);
  row_sumsq_k<<<dim3(MN), 128, 0, stream>>>(xn0, b2, 512);

  // Distance GEMM + softmax/one-hot aggregation, in query slabs
  float* out = (float*)d_out;
  for (int s = 0; s < MQ / QSLAB; ++s) {
    const __bf16* qs = xq0 + (size_t)s * QSLAB * 512;
    gemm_wmma_k<1><<<dim3(N_KEYS / TILE_N, QSLAB / TILE_M), 256, 0, stream>>>(
        qs, xn0, nullptr, a2 + (size_t)s * QSLAB, b2, nullptr, sim,
        QSLAB, N_KEYS, 512);
    softmax_agg_k<<<dim3(QSLAB), 256, 0, stream>>>(
        sim, yn, out + (size_t)s * QSLAB * N_CLS);
  }
}
